// CrossModalTranslator_9096740733589
// MI455X (gfx1250) — compile-verified, matrix operand order checked by reading, not run on hardware
//
#include <hip/hip_runtime.h>
#include <hip/hip_bf16.h>
#include <math.h>

typedef _Float16 v16h __attribute__((ext_vector_type(16)));
typedef _Float16 h8   __attribute__((ext_vector_type(8)));
typedef float    v8f  __attribute__((ext_vector_type(8)));

#define WMMA_F16(a, b, c) \
  __builtin_amdgcn_wmma_f32_16x16x32_f16(false, (a), false, (b), (short)0, (c), false, false)

__device__ __forceinline__ v16h cat8(h8 lo, h8 hi) {
  v16h r;
#pragma unroll
  for (int i = 0; i < 8; ++i) { r[i] = lo[i]; r[i + 8] = hi[i]; }
  return r;
}

// B operand, pre-laid-out by prep kernel: lane l holds 16 contiguous halfs.
__device__ __forceinline__ v16h load_b_frag(const _Float16* __restrict__ w, int tile, int lane) {
  const h8* q = (const h8*)(w + (((size_t)tile * 32 + lane) << 4));
  return cat8(q[0], q[1]);
}

// A operand from f32 global memory: 16 contiguous floats per lane -> f16.
__device__ __forceinline__ v16h load_a_f32(const float* __restrict__ p) {
  const float4* q = (const float4*)p;
  float4 f0 = q[0], f1 = q[1], f2 = q[2], f3 = q[3];
  v16h a;
  a[0]  = (_Float16)f0.x; a[1]  = (_Float16)f0.y; a[2]  = (_Float16)f0.z; a[3]  = (_Float16)f0.w;
  a[4]  = (_Float16)f1.x; a[5]  = (_Float16)f1.y; a[6]  = (_Float16)f1.z; a[7]  = (_Float16)f1.w;
  a[8]  = (_Float16)f2.x; a[9]  = (_Float16)f2.y; a[10] = (_Float16)f2.z; a[11] = (_Float16)f2.w;
  a[12] = (_Float16)f3.x; a[13] = (_Float16)f3.y; a[14] = (_Float16)f3.z; a[15] = (_Float16)f3.w;
  return a;
}

// sum across a 16-lane half (rows for lanes 0-15 differ from lanes 16-31)
__device__ __forceinline__ float redux16(float v) {
  v += __shfl_xor(v, 1, 16);
  v += __shfl_xor(v, 2, 16);
  v += __shfl_xor(v, 4, 16);
  v += __shfl_xor(v, 8, 16);
  return v;
}

__device__ __forceinline__ float geluf(float x) {
  return 0.5f * x * (1.0f + erff(0.70710678118654752f * x));
}

// In-register bias + LayerNorm + exact GELU over NT*16 columns.
// acc[nt][j] = element (row j + 8*hi, col nt*16 + ln) of the C tiles.
template <int NT>
__device__ __forceinline__ void bias_ln_gelu(v8f* acc, const float* __restrict__ bias,
                                             const float* __restrict__ gamma,
                                             const float* __restrict__ beta, int ln) {
  constexpr float invN = 1.0f / (float)(NT * 16);
  float bl[NT], gl[NT], el[NT];
#pragma unroll
  for (int nt = 0; nt < NT; ++nt) {
    int c = nt * 16 + ln;
    bl[nt] = bias[c]; gl[nt] = gamma[c]; el[nt] = beta[c];
  }
#pragma unroll
  for (int j = 0; j < 8; ++j) {
    float s = 0.f, q = 0.f;
#pragma unroll
    for (int nt = 0; nt < NT; ++nt) {
      float v = acc[nt][j] + bl[nt];
      acc[nt][j] = v; s += v; q += v * v;
    }
    s = redux16(s); q = redux16(q);
    float m = s * invN;
    float r = rsqrtf(q * invN - m * m + 1e-5f);
#pragma unroll
    for (int nt = 0; nt < NT; ++nt) {
      float v = (acc[nt][j] - m) * r * gl[nt] + el[nt];
      acc[nt][j] = geluf(v);
    }
  }
}

// ---------------------------------------------------------------------------
// Prep: rearrange f32 weights (Ktot x Ntot per p, P=6) into f16 WMMA B-frag
// order: half index = ((((p*KT+kt)*NT+nt)*32 + lane)*2 + c)*8 + i
// value = W[p][kt*32 + 16*(lane>>4) + 8*c + i][nt*16 + (lane&15)]
// ---------------------------------------------------------------------------
__global__ void xmt_prep_w(const float* __restrict__ W, _Float16* __restrict__ dst,
                           int KT, int NT, int Ktot, int Ntot, int total) {
  int t = blockIdx.x * blockDim.x + threadIdx.x;
  if (t >= total) return;
  int c  = t & 1;
  int l  = (t >> 1) & 31;
  int tn = t >> 6;
  int nt = tn % NT;
  int tk = tn / NT;
  int kt = tk % KT;
  int p  = tk / KT;
  int k0 = kt * 32 + 16 * (l >> 4) + 8 * c;
  int n  = nt * 16 + (l & 15);
  const float* s = W + ((size_t)p * Ktot + k0) * Ntot + n;
  h8 v;
#pragma unroll
  for (int i = 0; i < 8; ++i) v[i] = (_Float16)s[(size_t)i * Ntot];
  *(h8*)(dst + (size_t)t * 8) = v;
}

// ---------------------------------------------------------------------------
// Main fused kernel: 8 independent waves per block, 16 rows per wave.
// grid.x = P * (B / 128) = 6 * 128 = 768
// ---------------------------------------------------------------------------
__global__ __launch_bounds__(256) void xmt_main(
    const float* __restrict__ mus, const float* __restrict__ logvars,
    const unsigned char* __restrict__ mask,
    const float* __restrict__ hb1, const float* __restrict__ hg1, const float* __restrict__ hbe1,
    const float* __restrict__ hb2, const float* __restrict__ hg2, const float* __restrict__ hbe2,
    const float* __restrict__ hb3,
    const float* __restrict__ gb1, const float* __restrict__ gg1, const float* __restrict__ gbe1,
    const float* __restrict__ gw2, const float* __restrict__ gb2,
    const _Float16* __restrict__ whw1, const _Float16* __restrict__ whw2,
    const _Float16* __restrict__ whw3, const _Float16* __restrict__ wgw1,
    float* __restrict__ out) {
  constexpr int Bn = 16384;
  constexpr int XS = 104;  // padded LDS row stride in halfs (208B = 13*16B)
  __shared__ __align__(16) _Float16 xs[8][16 * XS];

  const int tid  = threadIdx.x;
  const int wv   = tid >> 5;
  const int lane = tid & 31;
  const int hi   = lane >> 4;
  const int ln   = lane & 15;

  const int blk     = blockIdx.x;
  const int p       = blk >> 7;  // / 128
  const int rowBase = (blk & 127) * 128 + wv * 16;

  const int src = p >> 1;                       // SRC = {0,0,1,1,2,2}
  const int tgt = (0x102021 >> (p * 4)) & 7;    // TGT = {1,2,0,2,0,1}

  // per-lane row mask (rows j + 8*hi) for the two output halves handled here
  bool mrow[8];
#pragma unroll
  for (int j = 0; j < 8; ++j) {
    int r = rowBase + j + 8 * hi;
    const unsigned char* mp = mask + (size_t)r * 3;
    mrow[j] = mp[src] && !mp[tgt];
  }
  const int ra = rowBase + ln;  // A-fragment row for this lane

  float* __restrict__ mu_out   = out;
  float* __restrict__ lv_out   = out + (size_t)6 * Bn * 256;
  float* __restrict__ gate_out = out + (size_t)12 * Bn * 256;

  // ---- GEMM1: h @ hw1 (K=512, N=96) fused with g1: h @ gw1 (N=32) ----
  const _Float16* w1p = whw1 + (size_t)p * (16 * 6 * 512);
  const _Float16* g1p = wgw1 + (size_t)p * (16 * 2 * 512);
  v8f accH[6] = {};
  v8f accG[2] = {};
  for (int kt = 0; kt < 16; ++kt) {
    const int col0 = kt * 32 + 16 * hi;  // multiple of 16; never crosses the 256 split
    const float* ap = (col0 < 256)
        ? (mus     + ((size_t)src * Bn + ra) * 256 + col0)
        : (logvars + ((size_t)src * Bn + ra) * 256 + (col0 - 256));
    v16h a = load_a_f32(ap);
#pragma unroll
    for (int nt = 0; nt < 6; ++nt)
      accH[nt] = WMMA_F16(a, load_b_frag(w1p, kt * 6 + nt, lane), accH[nt]);
#pragma unroll
    for (int nt = 0; nt < 2; ++nt)
      accG[nt] = WMMA_F16(a, load_b_frag(g1p, kt * 2 + nt, lane), accG[nt]);
  }

  bias_ln_gelu<6>(accH, hb1 + p * 96, hg1 + p * 96, hbe1 + p * 96, ln);
  // stage x as f16 in per-wave LDS slice (C-layout -> row-major)
#pragma unroll
  for (int nt = 0; nt < 6; ++nt)
#pragma unroll
    for (int j = 0; j < 8; ++j)
      xs[wv][(j + 8 * hi) * XS + nt * 16 + ln] = (_Float16)accH[nt][j];

  // ---- gate head: LN+GELU over 32 cols, dot with gw2, sigmoid ----
  bias_ln_gelu<2>(accG, gb1 + p * 32, gg1 + p * 32, gbe1 + p * 32, ln);
  {
    const float w0 = gw2[p * 32 + ln];
    const float w1 = gw2[p * 32 + 16 + ln];
    const float bb = gb2[p];
#pragma unroll
    for (int j = 0; j < 8; ++j) {
      float d = accG[0][j] * w0 + accG[1][j] * w1;
      d = redux16(d);
      if (ln == j) {  // lanes 0..7 store rows j, lanes 16..23 store rows j+8
        float gv = 1.0f / (1.0f + expf(-(d + bb)));
        gate_out[(size_t)p * Bn + rowBase + j + 8 * hi] = mrow[j] ? gv : 0.0f;
      }
    }
  }

  // ---- GEMM2: x @ hw2 (K=96, N=96) ----
  v16h a2[3];
#pragma unroll
  for (int kc = 0; kc < 3; ++kc) {
    const h8* q = (const h8*)&xs[wv][ln * XS + kc * 32 + 16 * hi];
    a2[kc] = cat8(q[0], q[1]);
  }
  const _Float16* w2p = whw2 + (size_t)p * (3 * 6 * 512);
  v8f accH2[6] = {};
#pragma unroll
  for (int nt = 0; nt < 6; ++nt)
#pragma unroll
    for (int kc = 0; kc < 3; ++kc)
      accH2[nt] = WMMA_F16(a2[kc], load_b_frag(w2p, kc * 6 + nt, lane), accH2[nt]);

  bias_ln_gelu<6>(accH2, hb2 + p * 96, hg2 + p * 96, hbe2 + p * 96, ln);
#pragma unroll
  for (int nt = 0; nt < 6; ++nt)
#pragma unroll
    for (int j = 0; j < 8; ++j)
      xs[wv][(j + 8 * hi) * XS + nt * 16 + ln] = (_Float16)accH2[nt][j];

  // ---- GEMM3: x2 @ hw3 (K=96, N=512) + bias, clip, mask, store ----
  v16h a3[3];
#pragma unroll
  for (int kc = 0; kc < 3; ++kc) {
    const h8* q = (const h8*)&xs[wv][ln * XS + kc * 32 + 16 * hi];
    a3[kc] = cat8(q[0], q[1]);
  }
  const _Float16* w3p = whw3 + (size_t)p * (3 * 32 * 512);
  const size_t rb = ((size_t)p * Bn + rowBase) * 256;
  for (int nt = 0; nt < 32; ++nt) {
    v8f acc = {};
#pragma unroll
    for (int kc = 0; kc < 3; ++kc)
      acc = WMMA_F16(a3[kc], load_b_frag(w3p, kc * 32 + nt, lane), acc);
    const int col  = nt * 16 + ln;
    const float b3 = hb3[p * 512 + col];
    const bool isMu = (col < 256);  // uniform across the wave per nt
    float* __restrict__ dst = isMu ? (mu_out + rb + col) : (lv_out + rb + col - 256);
#pragma unroll
    for (int j = 0; j < 8; ++j) {
      float v = acc[j] + b3;
      if (!isMu) v = fminf(fmaxf(v, -6.0f), 2.0f);
      v = mrow[j] ? v : 0.0f;
      dst[(size_t)(j + 8 * hi) * 256] = v;
    }
  }
}

extern "C" void kernel_launch(void* const* d_in, const int* in_sizes, int n_in,
                              void* d_out, int out_size, void* d_ws, size_t ws_size,
                              hipStream_t stream) {
  (void)in_sizes; (void)n_in; (void)out_size; (void)ws_size;
  const float* mus     = (const float*)d_in[0];
  const float* logvars = (const float*)d_in[1];
  const unsigned char* mask = (const unsigned char*)d_in[2];
  const float* hw1  = (const float*)d_in[3];
  const float* hb1  = (const float*)d_in[4];
  const float* hg1  = (const float*)d_in[5];
  const float* hbe1 = (const float*)d_in[6];
  const float* hw2  = (const float*)d_in[7];
  const float* hb2  = (const float*)d_in[8];
  const float* hg2  = (const float*)d_in[9];
  const float* hbe2 = (const float*)d_in[10];
  const float* hw3  = (const float*)d_in[11];
  const float* hb3  = (const float*)d_in[12];
  const float* gw1  = (const float*)d_in[13];
  const float* gb1  = (const float*)d_in[14];
  const float* gg1  = (const float*)d_in[15];
  const float* gbe1 = (const float*)d_in[16];
  const float* gw2  = (const float*)d_in[17];
  const float* gb2  = (const float*)d_in[18];
  float* out = (float*)d_out;

  _Float16* ws = (_Float16*)d_ws;
  // workspace layout (halfs)
  _Float16* whw1 = ws;                  // 6*512*96  = 294912
  _Float16* whw2 = ws + 294912;         // 6*96*96   = 55296
  _Float16* whw3 = ws + 350208;         // 6*96*512  = 294912
  _Float16* wgw1 = ws + 645120;         // 6*512*32  = 98304

  const int TH = 256;
  auto gl = [](int total) { return (total + 255) / 256; };
  // totals = P*KT*NT*64 (8-half groups)
  xmt_prep_w<<<gl(36864), TH, 0, stream>>>(hw1, whw1, 16, 6, 512, 96, 36864);
  xmt_prep_w<<<gl(6912),  TH, 0, stream>>>(hw2, whw2, 3, 6, 96, 96, 6912);
  xmt_prep_w<<<gl(36864), TH, 0, stream>>>(hw3, whw3, 3, 32, 96, 512, 36864);
  xmt_prep_w<<<gl(12288), TH, 0, stream>>>(gw1, wgw1, 16, 2, 512, 32, 12288);

  xmt_main<<<768, 256, 0, stream>>>(mus, logvars, mask,
                                    hb1, hg1, hbe1, hb2, hg2, hbe2, hb3,
                                    gb1, gg1, gbe1, gw2, gb2,
                                    whw1, whw2, whw3, wgw1, out);
}
